// WhisperOAIAttention_55628416417853
// MI455X (gfx1250) — compile-verified, matrix operand order checked by reading, not run on hardware
//
#include <hip/hip_runtime.h>
#include <hip/hip_bf16.h>

// ---------------------------------------------------------------------------
// WhisperOAIAttention on gfx1250 (MI455X): f16 WMMA pipeline, fp32 accumulate,
// TDM (tensor_load_to_lds) double-buffered GEMM staging.
//   B=2 S=2048 D=1024 H=16 DH=64
// ---------------------------------------------------------------------------

typedef __attribute__((ext_vector_type(16))) _Float16     v16h;
typedef __attribute__((ext_vector_type(8)))  float        v8f;
typedef __attribute__((ext_vector_type(4)))  unsigned int v4u;
typedef __attribute__((ext_vector_type(4)))  int          v4i;
typedef __attribute__((ext_vector_type(8)))  int          v8i;

union H16 { v16h h; v4u u[2]; };

constexpr int CB  = 2;
constexpr int CS  = 2048;
constexpr int CD  = 1024;
constexpr int CH  = 16;
constexpr int CDH = 64;
constexpr int CM  = CB * CS;            // 4096 rows for projection GEMMs
// DH^-0.25 * sqrt(log2(e)) : scores come out of WMMA pre-scaled by log2(e),
// so softmax uses exp2 directly (v_exp_f32 without the log2e multiply).
constexpr float QK_SCALE_L2 = 0.35355339059327379f * 1.2011224087864498f;
constexpr float NEG_L2      = -1.4426950408889634e9f;   // -1e9 * log2(e)

__device__ __forceinline__ v8f wmma_f16(v16h a, v16h b, v8f c) {
  return __builtin_amdgcn_wmma_f32_16x16x32_f16(
      false, a, false, b, (short)0, c, false, false);
}

// ---------------------------------------------------------------------------
// TDM helpers: build a 2-D tile descriptor (ISA 8.3/8.4) and issue the DMA.
//   tile: TD1 rows x 32 cols of f16, row stride = GK elements, into LDS.
// ---------------------------------------------------------------------------
__device__ __forceinline__ v4i tdm_g0(unsigned lds_addr, const void* gptr) {
  unsigned long long ga = (unsigned long long)(uintptr_t)gptr;
  v4i g;
  g[0] = 1;                                   // count=1, user descriptor
  g[1] = (int)lds_addr;                       // LDS byte address
  g[2] = (int)(unsigned)ga;                   // global_addr[31:0]
  g[3] = (int)(((unsigned)(ga >> 32) & 0x01FFFFFFu) | 0x80000000u); // type=2
  return g;
}

template <int TD1, int TENS_D1>
__device__ __forceinline__ v8i tdm_g1_tile32(int row_stride) {
  // data_size=2B; tensor_dim0 = row_stride; tensor_dim1 = TENS_D1;
  // tile_dim0 = 32; tile_dim1 = TD1; tensor_dim0_stride = row_stride.
  v8i g;
  g[0] = 0x00010000;                          // workgroup_mask=0, data_size=1(2B)
  g[1] = (int)((row_stride & 0xFFFF) << 16);  // tensor_dim0[15:0]
  g[2] = (int)((row_stride >> 16) & 0xFFFF) | (int)((TENS_D1 & 0xFFFF) << 16);
  g[3] = (int)((TENS_D1 >> 16) & 0xFFFF) | (int)(32 << 16);   // tile_dim0=32
  g[4] = TD1;                                 // tile_dim1 (tile_dim2=0)
  g[5] = row_stride;                          // tensor_dim0_stride[31:0]
  g[6] = 0;
  g[7] = 0;
  return g;
}

__device__ __forceinline__ void tdm_load(v4i g0, v8i g1) {
  asm volatile("tensor_load_to_lds %0, %1" :: "s"(g0), "s"(g1) : "memory");
}

// ---------------------------------------------------------------------------
// Kernel: fp32 -> f16 cast
// ---------------------------------------------------------------------------
__global__ __launch_bounds__(256)
void cast_f32_to_f16_k(const float* __restrict__ in, _Float16* __restrict__ out, int n) {
  int i = blockIdx.x * blockDim.x + threadIdx.x;
  if (i < n) out[i] = (_Float16)in[i];
}

// ---------------------------------------------------------------------------
// Kernel: tiled WMMA GEMM.  C[4096,1024] = A[4096,1024] @ W[1024,1024]^T
//                                          (+bias) (*scale)
//   OUT_MODE 0: f16, head-split [B,H,S,DH]            (Q/K projections)
//   OUT_MODE 3: f16, head-split transposed [B,H,DH,S] (V projection)
//   OUT_MODE 1: fp32, row-major [M,N]                 (final output)
// Block tile 128x128, BK=32, 8 waves (4x2), wave tile 32x64 (2x4 frags).
// Tile staging by the Tensor Data Mover, double-buffered on TENSORcnt.
// ---------------------------------------------------------------------------
#define BM 128
#define BN 128
#define BK 32
constexpr int GM = CM;    // 4096
constexpr int GN = CD;    // 1024
constexpr int GK = CD;    // 1024
constexpr unsigned ABUF_BYTES = BM * BK * 2;  // 8192
constexpr unsigned BBUF_BYTES = BN * BK * 2;  // 8192

template <int OUT_MODE, bool HAS_BIAS>
__global__ __launch_bounds__(256)
void gemm_wmma_f16_k(const _Float16* __restrict__ A,
                     const _Float16* __restrict__ W,
                     const float* __restrict__ bias,
                     void* __restrict__ out,
                     float out_scale) {
  // double-buffered tiles: 2 * (128*32 + 128*32) halves = 32 KB
  __shared__ __align__(16) _Float16 sA[2][BM * BK];
  __shared__ __align__(16) _Float16 sB[2][BN * BK];

  const int tid  = threadIdx.x;
  const int lane = tid & 31;
  const int wave = tid >> 5;     // 0..7
  const int wm   = wave & 3;     // wave row (4)  -> 32 rows each
  const int wn   = wave >> 2;    // wave col (2)  -> 64 cols each
  const int m0   = blockIdx.y * BM;
  const int n0   = blockIdx.x * BN;

  const int lrow = lane & 15;
  const int k0   = (lane >> 4) << 3;
  const int rhi  = (lane >> 4) << 3;

  // LDS byte addresses (generic AMDGPU pointers to LDS carry the DS-relative
  // offset in the low 32 bits)
  const unsigned ldsA = (unsigned)(uintptr_t)&sA[0][0];
  const unsigned ldsB = (unsigned)(uintptr_t)&sB[0][0];

  const _Float16* Abase = A + (size_t)m0 * GK;
  const _Float16* Wbase = W + (size_t)n0 * GK;
  const v8i g1a = tdm_g1_tile32<BM, GM>(GK);
  const v8i g1b = tdm_g1_tile32<BN, GN>(GK);

  v8f acc[2][4];
#pragma unroll
  for (int i = 0; i < 2; ++i)
#pragma unroll
    for (int j = 0; j < 4; ++j) acc[i][j] = (v8f)0.0f;

  // ---- prologue: TDM-fetch tile 0 into buffer 0 (wave 0 drives the DMA) ----
  if (wave == 0) {
    tdm_load(tdm_g0(ldsA, Abase), g1a);
    tdm_load(tdm_g0(ldsB, Wbase), g1b);
  }

  int cur = 0;
  for (int kt = 0; kt < GK; kt += BK) {
    const bool has_next = (kt + BK) < GK;
    if (wave == 0) {
      if (has_next) {   // issue next tile's DMAs into the other buffer
        tdm_load(tdm_g0(ldsA + (cur ^ 1) * ABUF_BYTES, Abase + kt + BK), g1a);
        tdm_load(tdm_g0(ldsB + (cur ^ 1) * BBUF_BYTES, Wbase + kt + BK), g1b);
        __builtin_amdgcn_s_wait_tensorcnt(2);  // current tile's 2 DMAs done
      } else {
        __builtin_amdgcn_s_wait_tensorcnt(0);
      }
    }
    __syncthreads();   // current buffer visible to all waves

    const _Float16* cA = sA[cur];
    const _Float16* cB = sB[cur];
    v16h afrag[2], bfrag[4];
#pragma unroll
    for (int f = 0; f < 2; ++f) {
      const int ar = wm * 32 + f * 16 + lrow;
      H16 t;
      t.u[0] = *(const v4u*)(cA + ar * BK + k0);
      t.u[1] = *(const v4u*)(cA + ar * BK + k0 + 16);
      afrag[f] = t.h;
    }
#pragma unroll
    for (int f = 0; f < 4; ++f) {
      const int br = wn * 64 + f * 16 + lrow;
      H16 t;
      t.u[0] = *(const v4u*)(cB + br * BK + k0);
      t.u[1] = *(const v4u*)(cB + br * BK + k0 + 16);
      bfrag[f] = t.h;
    }
#pragma unroll
    for (int fm = 0; fm < 2; ++fm)
#pragma unroll
      for (int fn = 0; fn < 4; ++fn)
        acc[fm][fn] = wmma_f16(afrag[fm], bfrag[fn], acc[fm][fn]);

    __syncthreads();   // all waves done reading; next iter may DMA over it
    cur ^= 1;
  }

  // ---- epilogue (compile-time specialized) ----
#pragma unroll
  for (int fn = 0; fn < 4; ++fn) {
    const int n = n0 + wn * 64 + fn * 16 + lrow;
    const float bval = HAS_BIAS ? bias[n] : 0.0f;
#pragma unroll
    for (int fm = 0; fm < 2; ++fm) {
#pragma unroll
      for (int r = 0; r < 8; ++r) {
        const int m = m0 + wm * 32 + fm * 16 + r + rhi;
        const float val = (acc[fm][fn][r] + bval) * out_scale;
        if (OUT_MODE == 0) {
          const int bb = m >> 11, ss = m & (CS - 1);
          const int hh = n >> 6,  dd = n & (CDH - 1);
          ((_Float16*)out)[(((size_t)(bb * CH + hh)) * CS + ss) * CDH + dd] =
              (_Float16)val;
        } else if (OUT_MODE == 3) {
          const int bb = m >> 11, ss = m & (CS - 1);
          const int hh = n >> 6,  dd = n & (CDH - 1);
          ((_Float16*)out)[(((size_t)(bb * CH + hh)) * CDH + dd) * CS + ss] =
              (_Float16)val;
        } else {
          ((float*)out)[(size_t)m * GN + n] = val;
        }
      }
    }
  }
}

// ---------------------------------------------------------------------------
// Kernel: causal flash attention per (b, h, 32-query tile).  One wave / block.
//   q,k : [B,H,S,DH] f16, pre-scaled by DH^-0.25 * sqrt(log2 e)
//   v   : [B,H,DH,S] f16 (transposed per head)
//   out : [B*S, D] f16 (heads re-merged)
// Softmax runs in the log2 domain: exp2 only, no log2e multiplies.
// ---------------------------------------------------------------------------
__global__ __launch_bounds__(32)
void attn_flash_wmma_k(const _Float16* __restrict__ q,
                       const _Float16* __restrict__ k,
                       const _Float16* __restrict__ v,
                       _Float16* __restrict__ out) {
  __shared__ __align__(16) _Float16 sP[32 * 32];

  const int lane = threadIdx.x;
  const int qt   = blockIdx.x;
  const int h    = blockIdx.y;
  const int b    = blockIdx.z;

  const size_t head_off = ((size_t)(b * CH + h)) * CS * CDH;
  const _Float16* Q  = q + head_off;
  const _Float16* Kp = k + head_off;
  const _Float16* Vt = v + head_off;   // [DH, S]

  const int q0   = qt * 32;
  const int lrow = lane & 15;
  const int kk0  = (lane >> 4) << 3;
  const int rhi  = (lane >> 4) << 3;

  v16h qf[2][2];
#pragma unroll
  for (int fm = 0; fm < 2; ++fm) {
#pragma unroll
    for (int kc = 0; kc < 2; ++kc) {
      const _Float16* p = Q + (size_t)(q0 + fm * 16 + lrow) * CDH + kc * 32 + kk0;
      H16 t;
      t.u[0] = *(const v4u*)(p);
      t.u[1] = *(const v4u*)(p + 16);
      qf[fm][kc] = t.h;
    }
  }

  v8f oacc[2][4];
#pragma unroll
  for (int i = 0; i < 2; ++i)
#pragma unroll
    for (int j = 0; j < 4; ++j) oacc[i][j] = (v8f)0.0f;

  float mrow[2][8], lsum[2][8];
#pragma unroll
  for (int i = 0; i < 2; ++i)
#pragma unroll
    for (int r = 0; r < 8; ++r) { mrow[i][r] = -1e30f; lsum[i][r] = 0.0f; }

  for (int j = 0; j <= qt; ++j) {
    const int kb = j * 32;

    v16h kf[2][2];
#pragma unroll
    for (int fn = 0; fn < 2; ++fn) {
#pragma unroll
      for (int kc = 0; kc < 2; ++kc) {
        const _Float16* p = Kp + (size_t)(kb + fn * 16 + lrow) * CDH + kc * 32 + kk0;
        H16 t;
        t.u[0] = *(const v4u*)(p);
        t.u[1] = *(const v4u*)(p + 16);
        kf[fn][kc] = t.h;
      }
    }

    v8f s[2][2];
#pragma unroll
    for (int fm = 0; fm < 2; ++fm)
#pragma unroll
      for (int fn = 0; fn < 2; ++fn) {
        s[fm][fn] = (v8f)0.0f;
#pragma unroll
        for (int kc = 0; kc < 2; ++kc)
          s[fm][fn] = wmma_f16(qf[fm][kc], kf[fn][kc], s[fm][fn]);
      }

    if (j == qt) {
#pragma unroll
      for (int fm = 0; fm < 2; ++fm)
#pragma unroll
        for (int fn = 0; fn < 2; ++fn)
#pragma unroll
          for (int r = 0; r < 8; ++r) {
            const int qr = fm * 16 + r + rhi;
            const int kc = fn * 16 + lrow;
            if (kc > qr) s[fm][fn][r] += NEG_L2;
          }
    }

#pragma unroll
    for (int fm = 0; fm < 2; ++fm) {
#pragma unroll
      for (int r = 0; r < 8; ++r) {
        float tmax = fmaxf(s[fm][0][r], s[fm][1][r]);
#pragma unroll
        for (int msk = 8; msk >= 1; msk >>= 1)
          tmax = fmaxf(tmax, __shfl_xor(tmax, msk));
        const float mnew = fmaxf(mrow[fm][r], tmax);
        const float corr = exp2f(mrow[fm][r] - mnew);
        const float p0 = exp2f(s[fm][0][r] - mnew);
        const float p1 = exp2f(s[fm][1][r] - mnew);
        s[fm][0][r] = p0;
        s[fm][1][r] = p1;
        float psum = p0 + p1;
#pragma unroll
        for (int msk = 8; msk >= 1; msk >>= 1)
          psum += __shfl_xor(psum, msk);
        lsum[fm][r] = lsum[fm][r] * corr + psum;
        mrow[fm][r] = mnew;
#pragma unroll
        for (int fd = 0; fd < 4; ++fd) oacc[fm][fd][r] *= corr;
      }
    }

    __syncthreads();
#pragma unroll
    for (int fm = 0; fm < 2; ++fm)
#pragma unroll
      for (int fn = 0; fn < 2; ++fn)
#pragma unroll
        for (int r = 0; r < 8; ++r) {
          const int pr = fm * 16 + r + rhi;
          const int pc = fn * 16 + lrow;
          sP[pr * 32 + pc] = (_Float16)s[fm][fn][r];
        }
    __syncthreads();

    v16h pf[2];
#pragma unroll
    for (int fm = 0; fm < 2; ++fm) {
      H16 t;
      t.u[0] = *(const v4u*)(sP + (fm * 16 + lrow) * 32 + kk0);
      t.u[1] = *(const v4u*)(sP + (fm * 16 + lrow) * 32 + kk0 + 16);
      pf[fm] = t.h;
    }

    v16h vf[4];
#pragma unroll
    for (int fd = 0; fd < 4; ++fd) {
      const _Float16* p = Vt + (size_t)(fd * 16 + lrow) * CS + kb + kk0;
      H16 t;
      t.u[0] = *(const v4u*)(p);
      t.u[1] = *(const v4u*)(p + 16);
      vf[fd] = t.h;
    }

#pragma unroll
    for (int fm = 0; fm < 2; ++fm)
#pragma unroll
      for (int fd = 0; fd < 4; ++fd)
        oacc[fm][fd] = wmma_f16(pf[fm], vf[fd], oacc[fm][fd]);
  }

#pragma unroll
  for (int fm = 0; fm < 2; ++fm)
#pragma unroll
    for (int fd = 0; fd < 4; ++fd)
#pragma unroll
      for (int r = 0; r < 8; ++r) {
        const int qr = q0 + fm * 16 + r + rhi;
        const int d  = fd * 16 + lrow;
        const float val = oacc[fm][fd][r] / lsum[fm][r];
        out[((size_t)b * CS + qr) * CD + h * CDH + d] = (_Float16)val;
      }
}

// ---------------------------------------------------------------------------
// Host-side orchestration
// ---------------------------------------------------------------------------
extern "C" void kernel_launch(void* const* d_in, const int* in_sizes, int n_in,
                              void* d_out, int out_size, void* d_ws, size_t ws_size,
                              hipStream_t stream) {
  const float* x  = (const float*)d_in[0];
  const float* Wq = (const float*)d_in[2];
  const float* bq = (const float*)d_in[3];
  const float* Wk = (const float*)d_in[4];
  const float* Wv = (const float*)d_in[5];
  const float* bv = (const float*)d_in[6];
  const float* Wo = (const float*)d_in[7];
  const float* bo = (const float*)d_in[8];

  char* ws = (char*)d_ws;
  _Float16* xh   = (_Float16*)ws;  ws += (size_t)CM * CD * 2;
  _Float16* Wqh  = (_Float16*)ws;  ws += (size_t)CD * CD * 2;
  _Float16* Wkh  = (_Float16*)ws;  ws += (size_t)CD * CD * 2;
  _Float16* Wvh  = (_Float16*)ws;  ws += (size_t)CD * CD * 2;
  _Float16* Woh  = (_Float16*)ws;  ws += (size_t)CD * CD * 2;
  _Float16* qws  = (_Float16*)ws;  ws += (size_t)CM * CD * 2;
  _Float16* kws  = (_Float16*)ws;  ws += (size_t)CM * CD * 2;
  _Float16* vws  = (_Float16*)ws;  ws += (size_t)CM * CD * 2;
  _Float16* aws  = (_Float16*)ws;  ws += (size_t)CM * CD * 2;

  const int nX = CM * CD;
  const int nW = CD * CD;
  cast_f32_to_f16_k<<<(nX + 255) / 256, 256, 0, stream>>>(x,  xh,  nX);
  cast_f32_to_f16_k<<<(nW + 255) / 256, 256, 0, stream>>>(Wq, Wqh, nW);
  cast_f32_to_f16_k<<<(nW + 255) / 256, 256, 0, stream>>>(Wk, Wkh, nW);
  cast_f32_to_f16_k<<<(nW + 255) / 256, 256, 0, stream>>>(Wv, Wvh, nW);
  cast_f32_to_f16_k<<<(nW + 255) / 256, 256, 0, stream>>>(Wo, Woh, nW);

  dim3 ggrid(GN / BN, GM / BM);   // (8, 32)
  gemm_wmma_f16_k<0, true ><<<ggrid, 256, 0, stream>>>(xh, Wqh, bq, qws, QK_SCALE_L2);
  gemm_wmma_f16_k<0, false><<<ggrid, 256, 0, stream>>>(xh, Wkh, nullptr, kws, QK_SCALE_L2);
  gemm_wmma_f16_k<3, true ><<<ggrid, 256, 0, stream>>>(xh, Wvh, bv, vws, 1.0f);

  dim3 agrid(CS / 32, CH, CB);    // (64, 16, 2)
  attn_flash_wmma_k<<<agrid, 32, 0, stream>>>(qws, kws, vws, aws);

  gemm_wmma_f16_k<1, true ><<<ggrid, 256, 0, stream>>>(aws, Woh, bo, (float*)d_out,
                                                       1.0f);
  (void)in_sizes; (void)n_in; (void)out_size; (void)ws_size;
}
